// SphericalHarmonicTransform_12223476924927
// MI455X (gfx1250) — matrix operand order repaired and checked by reading, not compiled, not run on hardware
//
#include <hip/hip_runtime.h>
#include <hip/hip_bf16.h>
#include <stdint.h>

// ---------------- problem constants (match reference) ----------------
#define BC      1024            // BATCH*CHAN
#define NLAT    192             // rings K
#define PMAX    404             // max nlon
#define P2      416             // PMAX padded to 13*32
#define MM      192             // zonal wavenumbers kept
#define LL      192             // degrees
#define NGRID   41088
#define TWO_PI_F 6.28318530717958647692f

#define PCH     32              // contraction chunk for 16x16x32 bf16 WMMA
#define NCH_A   13              // P2/32
#define NCH_B   6               // 192/32
#define LDSTR   40              // padded LDS row stride (bf16), mult of 8 for b128 align

typedef __attribute__((ext_vector_type(16))) __bf16       v16bf;
typedef __attribute__((ext_vector_type(8)))  float        v8f;
typedef __attribute__((ext_vector_type(4))) unsigned int  u32x4v;
typedef __attribute__((ext_vector_type(8)))  int          i32x8v;
typedef __attribute__((ext_vector_type(4)))  int          i32x4v;

union FragAB { v16bf v; u32x4v q[2]; };

__device__ __forceinline__ unsigned short bf_bits(float f) {
    __bf16 h = (__bf16)f;                      // hardware RNE cvt
    return __builtin_bit_cast(unsigned short, h);
}

// ----------------------------------------------------------------------------
// Tensor Data Mover: 2-D bf16 tile (32 elems per row) -> LDS with hardware row
// padding 64B + 16B (== bf16 stride 40, matching the WMMA fragment loads).
// D# per CDNA5 ISA ch.8: group0 = {count=1, lds_addr, global_addr, type=2};
// group1 = {data_size=2B, pad_enable, pad_interval=3 (16 DW), pad_amount=3 (4 DW),
//           tensor dims (huge: tiles never OOB), tile dims, dim0 stride}.
// ----------------------------------------------------------------------------
__device__ __forceinline__ void tdm_load_tile(unsigned lds_addr, const void* gptr,
                                              unsigned tile_d1, unsigned row_stride_elems) {
    const unsigned long long ga = (unsigned long long)(uintptr_t)gptr;
    u32x4v g0 = { 1u,
                  lds_addr,
                  (unsigned)ga,
                  (unsigned)((ga >> 32) & 0x1FFFFFFu) | (2u << 30) };
    i32x8v g1 = { (int)((1u << 16) | (1u << 20) | (3u << 22) | (3u << 25)),
                  0,
                  (int)0x4000u,
                  (int)(0x4000u | (32u << 16)),
                  (int)(tile_d1 & 0xFFFFu),
                  (int)row_stride_elems,
                  0, 0 };
    i32x4v z4 = { 0, 0, 0, 0 };
#if defined(__clang_major__) && (__clang_major__ >= 23)
    i32x8v z8 = { 0, 0, 0, 0, 0, 0, 0, 0 };
    __builtin_amdgcn_tensor_load_to_lds(g0, g1, z4, z4, z8, 0);
#else
    __builtin_amdgcn_tensor_load_to_lds(g0, g1, z4, z4, 0);
#endif
}

__device__ __forceinline__ unsigned lds_off(const void* p) {
    return (unsigned)(uintptr_t)p;             // flat LDS addr: low 32 bits = LDS offset
}

// ============================================================================
// Prep 1: basis -> bf16, 2pi-scaled, transposed to [k][m][p], p padded to 416.
// ============================================================================
__global__ __launch_bounds__(512)
void sht_prep_basis(const float* __restrict__ cosb, const float* __restrict__ sinb,
                    unsigned short* __restrict__ basis_t)   // [2][192][192][416]
{
    __shared__ float tc[32 * MM];
    __shared__ float ts[32 * MM];
    const int k = blockIdx.x, pt = blockIdx.y, tid = threadIdx.x;

    for (int i = tid; i < 32 * MM; i += 512) {            // m fastest: coalesced reads
        const int pl = i / MM, m = i - pl * MM;
        const int p = pt * 32 + pl;
        float cv = 0.f, sv = 0.f;
        if (p < PMAX) {
            const size_t g = ((size_t)(k * PMAX + p)) * MM + m;
            cv = TWO_PI_F * cosb[g];
            sv = TWO_PI_F * sinb[g];
        }
        tc[pl * MM + m] = cv;
        ts[pl * MM + m] = sv;
    }
    __syncthreads();
    for (int i = tid; i < MM * 32; i += 512) {            // p fastest: coalesced writes
        const int m = i >> 5, pl = i & 31;
        const size_t o = ((size_t)k * MM + m) * P2 + pt * 32 + pl;
        basis_t[o]                          = bf_bits(tc[pl * MM + m]);
        basis_t[(size_t)NLAT * MM * P2 + o] = bf_bits(ts[pl * MM + m]);
    }
}

// Prep 2: weight fp32 -> bf16 (layout (M,L,K) already B^T per m).
__global__ __launch_bounds__(512)
void sht_prep_weight(const float* __restrict__ w, unsigned short* __restrict__ wbf, int n)
{
    const int i = blockIdx.x * 512 + threadIdx.x;
    if (i < n) wbf[i] = bf_bits(w[i]);
}

// ============================================================================
// Stage A: per-ring masked DFT. re/im[bc][k][m] = sum_p xr[bc][p]*basis[k][p][m]
// grid.x = ring k (192), grid.y = bc-tile (8 x 128 rows), 512 threads = 16 waves.
// waves 0..7: cos -> re_ws ; 8..15: sin -> im_ws. Output bf16, layout [m][bc][k].
// 2-deep pipeline: basis via TDM ping-pong, x-gather prefetched into registers.
// ============================================================================
__global__ __launch_bounds__(512)
void sht_fourier_stage(const float* __restrict__ x,
                       const unsigned short* __restrict__ basis_t,
                       const int*   __restrict__ ring_idx,
                       unsigned short* __restrict__ re_ws,
                       unsigned short* __restrict__ im_ws)
{
    __shared__ unsigned short As[2][128 * LDSTR];     // gathered x, bf16
    __shared__ unsigned short Bb2[2][2][MM * LDSTR];  // [buf][cos|sin] basis [m][p]

    const int tid    = threadIdx.x;
    const int k      = blockIdx.x;
    const int bcbase = blockIdx.y * 128;
    const int wave   = tid >> 5;
    const int lane   = tid & 31;
    const int lh     = lane >> 4;
    const int ln     = lane & 15;
    const int hf     = wave >> 3;           // 0 = cos/re, 1 = sin/im
    const int rt     = wave & 7;

    const unsigned short* cos_t = basis_t + (size_t)k * MM * P2;
    const unsigned short* sin_t = cos_t + (size_t)NLAT * MM * P2;

    float xg[8];
    auto gather = [&](int pch) {            // 128x32 tile, 8 elems/thread -> regs
        #pragma unroll
        for (int j = 0; j < 8; ++j) {
            const int i = tid + j * 512;
            const int row = i >> 5, pl = i & 31;
            const int p = pch * PCH + pl;
            float v = 0.f;
            if (p < PMAX) {
                const int ridx = ring_idx[k * PMAX + p];
                v = x[(size_t)(bcbase + row) * NGRID + ridx];
            }
            xg[j] = v;
        }
    };
    auto stash = [&](int buf) {             // regs -> bf16 LDS
        #pragma unroll
        for (int j = 0; j < 8; ++j) {
            const int i = tid + j * 512;
            As[buf][(i >> 5) * LDSTR + (i & 31)] = bf_bits(xg[j]);
        }
    };
    auto issueB = [&](int buf, int pch) {   // 2 TDM tiles: 192 m rows x 32 p
        tdm_load_tile(lds_off(&Bb2[buf][0][0]), cos_t + pch * 32, MM, P2);
        tdm_load_tile(lds_off(&Bb2[buf][1][0]), sin_t + pch * 32, MM, P2);
    };

    // ---- pipeline preamble: chunk0 resident, chunk1 in flight ----
    gather(0);
    if (wave == 0) issueB(0, 0);
    stash(0);
    gather(1);
    if (wave == 0) {
        issueB(1, 1);
        __builtin_amdgcn_s_wait_tensorcnt(2);   // chunk0 basis complete (in-order)
    }
    __syncthreads();

    v8f acc[12] = {};
    for (int pch = 0; pch < NCH_A; ++pch) {
        const int cur = pch & 1, nxt = cur ^ 1;

        // A frag: K = 16*half+0..15 (matches B)
        FragAB a;
        const unsigned short* arow = &As[cur][(rt * 16 + ln) * LDSTR + 16 * lh];
        a.q[0] = *(const u32x4v*)(arow);
        a.q[1] = *(const u32x4v*)(arow + 8);

        const unsigned short* Bb = &Bb2[cur][hf][0];
        #pragma unroll
        for (int mt = 0; mt < 12; ++mt) {
            FragAB b;                          // B frag: N=lane%16, K=16*half+0..15
            const unsigned short* brow = &Bb[(mt * 16 + ln) * LDSTR + 16 * lh];
            b.q[0] = *(const u32x4v*)(brow);
            b.q[1] = *(const u32x4v*)(brow + 8);
            acc[mt] = __builtin_amdgcn_wmma_f32_16x16x32_bf16(
                false, a.v, false, b.v, (short)0, acc[mt], false, false);
        }

        if (pch + 1 < NCH_A) stash(nxt);       // chunk pch+1 regs -> alt buffer
        __syncthreads();                       // cur fully consumed, nxt stores visible
        if (pch + 2 < NCH_A) {
            if (wave == 0) issueB(cur, pch + 2);   // refill just-freed buffer
            gather(pch + 2);                       // prefetch gather into regs
            if (wave == 0) __builtin_amdgcn_s_wait_tensorcnt(2);  // pch+1 basis done
        } else if (pch + 1 < NCH_A) {
            if (wave == 0) __builtin_amdgcn_s_wait_tensorcnt(0);
        }
        if (pch + 1 < NCH_A) __syncthreads();
    }

    // ---- store bf16 to workspace, layout [m][bc][k] ----
    unsigned short* dst = hf ? im_ws : re_ws;
    #pragma unroll
    for (int mt = 0; mt < 12; ++mt) {
        #pragma unroll
        for (int d = 0; d < 8; ++d) {
            const int bc = bcbase + rt * 16 + d + 8 * lh;   // C/D: M = d (+8 hi half)
            const int m  = mt * 16 + ln;                    // N = lane%16
            dst[((size_t)m * BC + bc) * NLAT + k] = bf_bits(acc[mt][d]);
        }
    }
}

// ============================================================================
// Stage B: Legendre contraction. out[bc][l][m][{re,im}] = sum_k ws[m][bc][k]*w[m][l][k]
// grid.x = m (192), grid.y = bc-tile (8), 512 threads = 16 waves.
// All staged tiles are strided bf16 copies -> pure 2-deep TDM pipeline.
// ============================================================================
__global__ __launch_bounds__(512)
void sht_legendre_stage(const unsigned short* __restrict__ re_ws,
                        const unsigned short* __restrict__ im_ws,
                        const unsigned short* __restrict__ wbf,
                        float* __restrict__ out)
{
    __shared__ unsigned short Are[2][128 * LDSTR];
    __shared__ unsigned short Aim[2][128 * LDSTR];
    __shared__ unsigned short Bw [2][LL  * LDSTR];

    const int tid    = threadIdx.x;
    const int m      = blockIdx.x;
    const int bcbase = blockIdx.y * 128;
    const int wave   = tid >> 5;
    const int lane   = tid & 31;
    const int lh     = lane >> 4;
    const int ln     = lane & 15;
    const int hf     = wave >> 3;            // 0 = re->rl, 1 = im->il
    const int rt     = wave & 7;

    const unsigned short* reb = re_ws + ((size_t)m * BC + bcbase) * NLAT;
    const unsigned short* imb = im_ws + ((size_t)m * BC + bcbase) * NLAT;
    const unsigned short* wb  = wbf + (size_t)m * LL * NLAT;

    auto issue = [&](int buf, int kc) {      // 3 TDM tiles per chunk
        tdm_load_tile(lds_off(&Are[buf][0]), reb + kc * PCH, 128, NLAT);
        tdm_load_tile(lds_off(&Aim[buf][0]), imb + kc * PCH, 128, NLAT);
        tdm_load_tile(lds_off(&Bw [buf][0]), wb  + kc * PCH, LL,  NLAT);
    };

    if (wave == 0) {
        issue(0, 0);
        issue(1, 1);
        __builtin_amdgcn_s_wait_tensorcnt(3);    // chunk0's 3 tiles complete (in-order)
    }
    __syncthreads();

    v8f acc[12] = {};
    for (int kc = 0; kc < NCH_B; ++kc) {
        const int cur = kc & 1;

        FragAB a;
        const unsigned short* Ab = hf ? &Aim[cur][0] : &Are[cur][0];
        const unsigned short* arow = &Ab[(rt * 16 + ln) * LDSTR + 16 * lh];
        a.q[0] = *(const u32x4v*)(arow);
        a.q[1] = *(const u32x4v*)(arow + 8);

        #pragma unroll
        for (int lt = 0; lt < 12; ++lt) {
            FragAB b;
            const unsigned short* brow = &Bw[cur][(lt * 16 + ln) * LDSTR + 16 * lh];
            b.q[0] = *(const u32x4v*)(brow);
            b.q[1] = *(const u32x4v*)(brow + 8);
            acc[lt] = __builtin_amdgcn_wmma_f32_16x16x32_bf16(
                false, a.v, false, b.v, (short)0, acc[lt], false, false);
        }

        __syncthreads();                       // cur fully consumed
        if (kc + 2 < NCH_B) {
            if (wave == 0) {
                issue(cur, kc + 2);            // refill just-freed buffer
                __builtin_amdgcn_s_wait_tensorcnt(3);   // chunk kc+1 resident
            }
        } else if (kc + 1 < NCH_B) {
            if (wave == 0) __builtin_amdgcn_s_wait_tensorcnt(0);
        }
        if (kc + 1 < NCH_B) __syncthreads();
    }

    // out shape (B, C, L, M, 2), fp32
    #pragma unroll
    for (int lt = 0; lt < 12; ++lt) {
        #pragma unroll
        for (int d = 0; d < 8; ++d) {
            const int bc = bcbase + rt * 16 + d + 8 * lh;
            const int l  = lt * 16 + ln;
            out[(((size_t)bc * LL + l) * MM + m) * 2 + hf] = acc[lt][d];
        }
    }
}

// ============================================================================
extern "C" void kernel_launch(void* const* d_in, const int* in_sizes, int n_in,
                              void* d_out, int out_size, void* d_ws, size_t ws_size,
                              hipStream_t stream) {
    const float* x        = (const float*)d_in[0];
    const float* weight   = (const float*)d_in[1];
    const float* cosb     = (const float*)d_in[2];
    const float* sinb     = (const float*)d_in[3];
    const int*   ring_idx = (const int*)d_in[4];
    float*       out      = (float*)d_out;

    // workspace carve-up (bf16 elements):
    //   re_ws / im_ws : [m][bc][k]     -> 2 * 192*1024*192
    //   basis_t       : [2][k][m][416] -> 2 * 192*192*416
    //   weight_bf     : [m][l][k]      ->     192*192*192
    unsigned short* re_ws     = (unsigned short*)d_ws;
    unsigned short* im_ws     = re_ws  + (size_t)MM * BC * NLAT;
    unsigned short* basis_t   = im_ws  + (size_t)MM * BC * NLAT;
    unsigned short* weight_bf = basis_t + (size_t)2 * NLAT * MM * P2;

    dim3 blk(512, 1, 1);
    sht_prep_basis<<<dim3(NLAT, NCH_A, 1), blk, 0, stream>>>(cosb, sinb, basis_t);
    const int nw = MM * LL * NLAT;
    sht_prep_weight<<<dim3((nw + 511) / 512, 1, 1), blk, 0, stream>>>(weight, weight_bf, nw);
    sht_fourier_stage<<<dim3(NLAT, BC / 128, 1), blk, 0, stream>>>(
        x, basis_t, ring_idx, re_ws, im_ws);
    sht_legendre_stage<<<dim3(MM, BC / 128, 1), blk, 0, stream>>>(
        re_ws, im_ws, weight_bf, out);
}
